// BetaSample_33535104647650
// MI455X (gfx1250) — matrix operand order repaired and checked by reading, not compiled, not run on hardware
//
#include <hip/hip_runtime.h>
#include <hip/hip_bf16.h>
#include <stdint.h>

// CDNA5 WMMA vector types
typedef __attribute__((ext_vector_type(16))) __bf16 v16bf;
typedef __attribute__((ext_vector_type(8)))  float  v8f;
// TDM descriptor vector types (per CDNA5_HIP.md probe table)
typedef unsigned int uint32x4 __attribute__((ext_vector_type(4)));
typedef int          int32x4  __attribute__((ext_vector_type(4)));
typedef int          int32x8  __attribute__((ext_vector_type(8)));

#define BM 128           // rows of x per workgroup
#define BN 128           // output features per workgroup
#define BK 32            // K per WMMA step (bf16 16x16x32)
#define LDSS 40          // shorts per LDS row (80 B: 16B-aligned, conflict-free)
#define EPW 132          // per-thread floats in LDS epilogue buffer (528 B, 16B-aligned)

#if __has_builtin(__builtin_amdgcn_tensor_load_to_lds) && __has_builtin(__builtin_amdgcn_s_wait_tensorcnt)
#define HAVE_TDM 1
#else
#define HAVE_TDM 0
#endif

// ---- fp32 -> packed bf16 pair with RNE, using one v_perm_b32 for the merge ----
__device__ __forceinline__ unsigned pack_bf16(float x, float y) {
  union { float f; unsigned u; } ux, uy;
  ux.f = x; uy.f = y;
  unsigned rx = ux.u + 0x7FFFu + ((ux.u >> 16) & 1u);
  unsigned ry = uy.u + 0x7FFFu + ((uy.u >> 16) & 1u);
  return __builtin_amdgcn_perm(ry, rx, 0x07060302u);
}

// ---- deterministic counter-hash RNG (PCG-ish), per-element seeded ----
__device__ __forceinline__ unsigned rng_next(unsigned &s) {
  s = s * 747796405u + 2891336453u;
  unsigned w = ((s >> ((s >> 28) + 4u)) ^ s) * 277803737u;
  return (w >> 22) ^ w;
}
__device__ __forceinline__ float rng_uniform(unsigned &s) {
  return ((float)(rng_next(s) >> 8) + 0.5f) * (1.0f / 16777216.0f);
}
__device__ __forceinline__ float rng_normal(unsigned &s) {
  float u1 = rng_uniform(s), u2 = rng_uniform(s);
  return __fsqrt_rn(-2.0f * __logf(u1)) * __cosf(6.28318530718f * u2);
}
// Marsaglia–Tsang Gamma(alpha) sampler, bounded rejection loop (deterministic)
__device__ float gamma_sample(float alpha, unsigned &s) {
  float boost = 1.0f;
  if (alpha < 1.0f) {
    boost = __powf(rng_uniform(s), 1.0f / alpha);
    alpha += 1.0f;
  }
  float d = alpha - (1.0f / 3.0f);
  float c = 1.0f / __fsqrt_rn(9.0f * d);
  #pragma unroll 1
  for (int i = 0; i < 8; ++i) {
    float xn = rng_normal(s);
    float t = 1.0f + c * xn;
    if (t <= 0.0f) continue;
    float v = t * t * t;
    float u = rng_uniform(s);
    if (__logf(u) < 0.5f * xn * xn + d - d * v + d * __logf(v))
      return boost * d * v;
  }
  return boost * d;
}

// ---- fused epilogue: stage accumulators through LDS (NOT scratch).
// Store phase is fully unrolled (constant vector extracts -> stays in VGPRs,
// ds_store_b128). Compute phase uses dynamic LDS addressing (ds_load_b32),
// which is cheap, unlike dynamic VGPR-array indexing that forced scratch
// spills of all 128 accumulators in the previous round.
__device__ __forceinline__ void beta_epilogue(
    v8f (&acc_a)[4][2], v8f (&acc_b)[4][2], float* sEpi,
    int m0, int n0, int wm, int wn, int laneHi, int lr,
    const float* __restrict__ X, const float* __restrict__ ba,
    const float* __restrict__ bb, float* __restrict__ out, int D) {
  const int base = threadIdx.x * EPW;
  #pragma unroll
  for (int mt = 0; mt < 4; ++mt) {
    #pragma unroll
    for (int nt = 0; nt < 2; ++nt) {
      const int t = mt * 2 + nt;
      union { v8f v; float4 q[2]; } ua, ub;
      ua.v = acc_a[mt][nt];
      ub.v = acc_b[mt][nt];
      *(float4*)&sEpi[base + t * 8]          = ua.q[0];
      *(float4*)&sEpi[base + t * 8 + 4]      = ua.q[1];
      *(float4*)&sEpi[base + 64 + t * 8]     = ub.q[0];
      *(float4*)&sEpi[base + 64 + t * 8 + 4] = ub.q[1];
    }
  }
  #pragma unroll 1
  for (int idx = 0; idx < 64; ++idx) {
    const int e  = idx & 7;
    const int t  = idx >> 3;
    const int nt = t & 1;
    const int mt = t >> 1;
    const float va = sEpi[base + idx];
    const float vb = sEpi[base + 64 + idx];
    const int gi = m0 + wm * 64 + mt * 16 + e + laneHi * 8;  // C/D: M = vgpr + 8*half
    const int gj = n0 + wn * 32 + nt * 16 + lr;              // N = lane&15
    float a = va + ba[gj];
    float z = vb + bb[gj];
    float b = (z > 20.0f) ? z : log1pf(__expf(z));           // softplus
    a = fmaxf(a, 1e-6f);
    b = fmaxf(b, 1e-6f);
    const float xv = X[(size_t)gi * D + gj];
    unsigned st = ((unsigned)(gi * D + gj)) * 2654435761u ^ 0x9E3779B9u;
    const float gab = gamma_sample(a + b, st);
    const float ga  = gamma_sample(a, st);
    const float gb  = gamma_sample(b, st);
    const float pdf = gab * __powf(xv, a - 1.0f) * __powf(1.0f - xv, b - 1.0f) / (ga + gb);
    out[(size_t)gi * D + gj] = pdf;
  }
}

// =====================================================================
// Kernel 1: fp32 -> bf16 pre-pass (grid-stride, float4 -> uint2)
// =====================================================================
__global__ __launch_bounds__(256)
void cvt_bf16_kernel(const float* __restrict__ src, unsigned* __restrict__ dst, long n4) {
  long i = (long)blockIdx.x * blockDim.x + threadIdx.x;
  long stride = (long)gridDim.x * blockDim.x;
  for (; i < n4; i += stride) {
    float4 f = ((const float4*)src)[i];
    uint2 p;
    p.x = pack_bf16(f.x, f.y);
    p.y = pack_bf16(f.z, f.w);
    ((uint2*)dst)[i] = p;
  }
}

// =====================================================================
// TDM descriptor issue: 2D tile (32 x 128 bf16) with LDS padding so rows
// land at 80 B stride (16-dword interval, 4-dword pad) per ISA 08 §8.
// =====================================================================
#if HAVE_TDM
__device__ __forceinline__ void tdm_load_tile(unsigned lds_addr, unsigned long long ga,
                                              int tdim0, int tdim1, int stride0) {
  uint32x4 g0;
  g0[0] = 1u;                                                 // count=1, user desc
  g0[1] = lds_addr;                                           // LDS byte address
  g0[2] = (unsigned)ga;                                       // global_addr[31:0]
  g0[3] = (unsigned)((ga >> 32) & 0x1FFFFFFull) | (2u << 30); // addr[56:32] | type=2
  int32x8 g1;
  g1[0] = (int)((1u << 16)      // data_size = 2 bytes
              | (1u << 20)      // pad_enable
              | (3u << 22)      // pad_interval: 16 dwords (one 64 B row)
              | (3u << 25));    // pad_amount: 4 dwords (16 B) -> 80 B LDS stride
  g1[1] = (int)(((unsigned)tdim0 & 0xFFFFu) << 16);
  g1[2] = (int)(((unsigned)tdim0 >> 16) | (((unsigned)tdim1 & 0xFFFFu) << 16));
  g1[3] = (int)(((unsigned)tdim1 >> 16) | (32u << 16));       // tile_dim0 = 32
  g1[4] = 128;                                                // tile_dim1 = 128
  g1[5] = stride0;                                            // tensor_dim0_stride lo
  g1[6] = 0;
  g1[7] = 0;
  int32x4 z4 = {0, 0, 0, 0};
#if defined(__clang_major__) && (__clang_major__ >= 23)
  int32x8 z8 = {0, 0, 0, 0, 0, 0, 0, 0};
  __builtin_amdgcn_tensor_load_to_lds(g0, g1, z4, z4, z8, 0);
#else
  __builtin_amdgcn_tensor_load_to_lds(g0, g1, z4, z4, 0);
#endif
}
#endif

// =====================================================================
// Kernel 2: bf16-workspace GEMM, TDM-staged (double buffered) + epilogue
// =====================================================================
__global__ __launch_bounds__(256)
void beta_gemm_bf16(const unsigned short* __restrict__ Xb,
                    const unsigned short* __restrict__ Ab,
                    const unsigned short* __restrict__ Bb,
                    const float* __restrict__ X,  const float* __restrict__ ba,
                    const float* __restrict__ bb, float* __restrict__ out,
                    int D, int Bsz) {
  __shared__ unsigned short sX[2][BM * LDSS];
  __shared__ unsigned short sA[2][BN * LDSS];
  __shared__ unsigned short sB[2][BN * LDSS];
  __shared__ float sEpi[256 * EPW];

  const int tid    = threadIdx.x;
  const int wave   = tid >> 5;
  const int lane   = tid & 31;
  const int wm     = wave & 1;
  const int wn     = wave >> 1;
  const int laneHi = lane >> 4;
  const int lr     = lane & 15;

  const int m0 = blockIdx.x * BM;   // M-major walk keeps W panel hot in L2
  const int n0 = blockIdx.y * BN;

  v8f acc_a[4][2], acc_b[4][2];
  #pragma unroll
  for (int mt = 0; mt < 4; ++mt)
    #pragma unroll
    for (int nt = 0; nt < 2; ++nt)
      #pragma unroll
      for (int e = 0; e < 8; ++e) { acc_a[mt][nt][e] = 0.0f; acc_b[mt][nt][e] = 0.0f; }

  const int kTiles = D / BK;

#if HAVE_TDM
  const unsigned long long gx  = (unsigned long long)(uintptr_t)(Xb + (size_t)m0 * D);
  const unsigned long long gaw = (unsigned long long)(uintptr_t)(Ab + (size_t)n0 * D);
  const unsigned long long gbw = (unsigned long long)(uintptr_t)(Bb + (size_t)n0 * D);
  if (wave == 0) {   // TDM ignores EXEC; issue once per workgroup (wave 0)
    tdm_load_tile((unsigned)(uintptr_t)&sX[0][0], gx,  D, Bsz, D);
    tdm_load_tile((unsigned)(uintptr_t)&sA[0][0], gaw, D, D,   D);
    tdm_load_tile((unsigned)(uintptr_t)&sB[0][0], gbw, D, D,   D);
    __builtin_amdgcn_s_wait_tensorcnt(0);
  }
  __syncthreads();
#endif

  for (int kt = 0; kt < kTiles; ++kt) {
    const int cur = kt & 1;
#if HAVE_TDM
    // issue next tile's DMA into the other buffer, overlapping with WMMA
    if (wave == 0 && (kt + 1) < kTiles) {
      const unsigned long long koff = (unsigned long long)(kt + 1) * BK * 2ull;
      tdm_load_tile((unsigned)(uintptr_t)&sX[cur ^ 1][0], gx  + koff, D, Bsz, D);
      tdm_load_tile((unsigned)(uintptr_t)&sA[cur ^ 1][0], gaw + koff, D, D,   D);
      tdm_load_tile((unsigned)(uintptr_t)&sB[cur ^ 1][0], gbw + koff, D, D,   D);
    }
#else
    {
      const int k0 = kt * BK;
      #pragma unroll
      for (int i = 0; i < 2; ++i) {
        int idx = tid + i * 256;
        int row = idx >> 3, c4 = idx & 7;
        *(uint2*)&sX[cur][row * LDSS + c4 * 4] =
            *(const uint2*)(Xb + (size_t)(m0 + row) * D + k0 + c4 * 4);
        *(uint2*)&sA[cur][row * LDSS + c4 * 4] =
            *(const uint2*)(Ab + (size_t)(n0 + row) * D + k0 + c4 * 4);
        *(uint2*)&sB[cur][row * LDSS + c4 * 4] =
            *(const uint2*)(Bb + (size_t)(n0 + row) * D + k0 + c4 * 4);
      }
      if (kt + 1 < kTiles)
        __builtin_prefetch(Xb + (size_t)(m0 + (tid >> 1)) * D + k0 + BK, 0, 0);
      __syncthreads();
    }
#endif

    // ---- A fragments ----
    v16bf xf[4];
    #pragma unroll
    for (int mt = 0; mt < 4; ++mt) {
      const unsigned short* p = &sX[cur][(wm * 64 + mt * 16 + lr) * LDSS + laneHi * 16];
      union { uint4 q[2]; v16bf v; } u;
      u.q[0] = *(const uint4*)p;
      u.q[1] = *(const uint4*)(p + 8);
      xf[mt] = u.v;
    }
    // ---- B fragments + 16 WMMAs ----
    #pragma unroll
    for (int nt = 0; nt < 2; ++nt) {
      const unsigned short* pa = &sA[cur][(wn * 32 + nt * 16 + lr) * LDSS + laneHi * 16];
      union { uint4 q[2]; v16bf v; } ua;
      ua.q[0] = *(const uint4*)pa; ua.q[1] = *(const uint4*)(pa + 8);
      const unsigned short* pb = &sB[cur][(wn * 32 + nt * 16 + lr) * LDSS + laneHi * 16];
      union { uint4 q[2]; v16bf v; } ub;
      ub.q[0] = *(const uint4*)pb; ub.q[1] = *(const uint4*)(pb + 8);
      #pragma unroll
      for (int mt = 0; mt < 4; ++mt) {
        acc_a[mt][nt] = __builtin_amdgcn_wmma_f32_16x16x32_bf16(
            false, xf[mt], false, ua.v, (short)0, acc_a[mt][nt], false, false);
        acc_b[mt][nt] = __builtin_amdgcn_wmma_f32_16x16x32_bf16(
            false, xf[mt], false, ub.v, (short)0, acc_b[mt][nt], false, false);
      }
    }
#if HAVE_TDM
    if (wave == 0 && (kt + 1) < kTiles) __builtin_amdgcn_s_wait_tensorcnt(0);
#endif
    __syncthreads();
  }

  beta_epilogue(acc_a, acc_b, sEpi, m0, n0, wm, wn, laneHi, lr, X, ba, bb, out, D);
}

// =====================================================================
// Kernel 3: fully fused fallback (fp32 inputs, in-loop bf16 conversion)
// used only when d_ws cannot hold the bf16 workspace.
// =====================================================================
__global__ __launch_bounds__(256)
void beta_pdf_fused(const float* __restrict__ X,  const float* __restrict__ Wa,
                    const float* __restrict__ ba, const float* __restrict__ Wb,
                    const float* __restrict__ bb, float* __restrict__ out,
                    int D) {
  __shared__ unsigned short sX[BM * LDSS];
  __shared__ unsigned short sA[BN * LDSS];
  __shared__ unsigned short sB[BN * LDSS];
  __shared__ float sEpi[256 * EPW];

  const int tid    = threadIdx.x;
  const int wave   = tid >> 5;
  const int lane   = tid & 31;
  const int wm     = wave & 1;
  const int wn     = wave >> 1;
  const int laneHi = lane >> 4;
  const int lr     = lane & 15;
  const int m0 = blockIdx.x * BM;
  const int n0 = blockIdx.y * BN;

  v8f acc_a[4][2], acc_b[4][2];
  #pragma unroll
  for (int mt = 0; mt < 4; ++mt)
    #pragma unroll
    for (int nt = 0; nt < 2; ++nt)
      #pragma unroll
      for (int e = 0; e < 8; ++e) { acc_a[mt][nt][e] = 0.0f; acc_b[mt][nt][e] = 0.0f; }

  const int kTiles = D / BK;
  for (int kt = 0; kt < kTiles; ++kt) {
    const int k0 = kt * BK;
    #pragma unroll
    for (int i = 0; i < 4; ++i) {
      int idx = tid + i * 256;
      int row = idx >> 3, c4 = idx & 7;
      float4 f = *(const float4*)(X + (size_t)(m0 + row) * D + k0 + c4 * 4);
      uint2 p; p.x = pack_bf16(f.x, f.y); p.y = pack_bf16(f.z, f.w);
      *(uint2*)&sX[row * LDSS + c4 * 4] = p;
      f = *(const float4*)(Wa + (size_t)(n0 + row) * D + k0 + c4 * 4);
      p.x = pack_bf16(f.x, f.y); p.y = pack_bf16(f.z, f.w);
      *(uint2*)&sA[row * LDSS + c4 * 4] = p;
      f = *(const float4*)(Wb + (size_t)(n0 + row) * D + k0 + c4 * 4);
      p.x = pack_bf16(f.x, f.y); p.y = pack_bf16(f.z, f.w);
      *(uint2*)&sB[row * LDSS + c4 * 4] = p;
    }
    if (kt + 1 < kTiles)
      __builtin_prefetch(X + (size_t)(m0 + (tid >> 1)) * D + (k0 + BK) + (tid & 1) * 16, 0, 0);
    __syncthreads();

    v16bf xf[4];
    #pragma unroll
    for (int mt = 0; mt < 4; ++mt) {
      const unsigned short* p = &sX[(wm * 64 + mt * 16 + lr) * LDSS + laneHi * 16];
      union { uint4 q[2]; v16bf v; } u;
      u.q[0] = *(const uint4*)p; u.q[1] = *(const uint4*)(p + 8);
      xf[mt] = u.v;
    }
    #pragma unroll
    for (int nt = 0; nt < 2; ++nt) {
      const unsigned short* pa = &sA[(wn * 32 + nt * 16 + lr) * LDSS + laneHi * 16];
      union { uint4 q[2]; v16bf v; } ua;
      ua.q[0] = *(const uint4*)pa; ua.q[1] = *(const uint4*)(pa + 8);
      const unsigned short* pb = &sB[(wn * 32 + nt * 16 + lr) * LDSS + laneHi * 16];
      union { uint4 q[2]; v16bf v; } ub;
      ub.q[0] = *(const uint4*)pb; ub.q[1] = *(const uint4*)(pb + 8);
      #pragma unroll
      for (int mt = 0; mt < 4; ++mt) {
        acc_a[mt][nt] = __builtin_amdgcn_wmma_f32_16x16x32_bf16(
            false, xf[mt], false, ua.v, (short)0, acc_a[mt][nt], false, false);
        acc_b[mt][nt] = __builtin_amdgcn_wmma_f32_16x16x32_bf16(
            false, xf[mt], false, ub.v, (short)0, acc_b[mt][nt], false, false);
      }
    }
    __syncthreads();
  }

  beta_epilogue(acc_a, acc_b, sEpi, m0, n0, wm, wn, laneHi, lr, X, ba, bb, out, D);
}

extern "C" void kernel_launch(void* const* d_in, const int* in_sizes, int n_in,
                              void* d_out, int out_size, void* d_ws, size_t ws_size,
                              hipStream_t stream) {
  const float* X  = (const float*)d_in[0];
  const float* Wa = (const float*)d_in[1];
  const float* ba = (const float*)d_in[2];
  const float* Wb = (const float*)d_in[3];
  const float* bb = (const float*)d_in[4];
  float* out = (float*)d_out;
  const int D   = in_sizes[2];          // 4096
  const int Bsz = in_sizes[0] / D;      // 8192

  const size_t nX = (size_t)Bsz * D;
  const size_t nW = (size_t)D * D;
  const size_t need = (nX + 2 * nW) * sizeof(unsigned short);

  dim3 grid(Bsz / BM, D / BN);          // M-major: W panel stays hot in L2

  if (ws_size >= need) {
    unsigned short* wsX = (unsigned short*)d_ws;
    unsigned short* wsA = wsX + nX;
    unsigned short* wsB = wsA + nW;
    hipLaunchKernelGGL(cvt_bf16_kernel, dim3(4096), dim3(256), 0, stream,
                       X,  (unsigned*)wsX, (long)(nX / 4));
    hipLaunchKernelGGL(cvt_bf16_kernel, dim3(4096), dim3(256), 0, stream,
                       Wa, (unsigned*)wsA, (long)(nW / 4));
    hipLaunchKernelGGL(cvt_bf16_kernel, dim3(4096), dim3(256), 0, stream,
                       Wb, (unsigned*)wsB, (long)(nW / 4));
    hipLaunchKernelGGL(beta_gemm_bf16, grid, dim3(256), 0, stream,
                       wsX, wsA, wsB, X, ba, bb, out, D, Bsz);
  } else {
    hipLaunchKernelGGL(beta_pdf_fused, grid, dim3(256), 0, stream,
                       X, Wa, ba, Wb, bb, out, D);
  }
}
